// DSNet_9792525435438
// MI455X (gfx1250) — matrix operand order repaired and checked by reading, not compiled, not run on hardware
//
#include <hip/hip_runtime.h>
#include <hip/hip_bf16.h>

// ---------------- CDNA5 WMMA types ----------------
typedef __attribute__((ext_vector_type(16))) __bf16 v16bf;
typedef __attribute__((ext_vector_type(8)))  float  v8f;

union BfVec {
    v16bf v;
    unsigned int u[8];
    uint4 u4[2];
};

__device__ __forceinline__ unsigned short f2bf(float f) {
    union { __hip_bfloat16 b; unsigned short u; } cv;
    cv.b = __float2bfloat16(f);      // RNE, single v_cvt on gfx1250
    return cv.u;
}

#define NB   8
#define CIN  64
#define IHH  384
#define IWW  384
#define OH   192
#define OW   192
#define PLANE_IN  (IHH*IWW)
#define PLANE_OUT (OH*OW)

// channel-shuffle(g=8) on the 128-channel concat: oc = (ic%16)*8 + ic/16
__device__ __forceinline__ int shuffled(int ic) { return ((ic & 15) << 3) + (ic >> 4); }

// ---------------- 0: zero BN stat accumulators ----------------
__global__ void zero_stats_k(float* s) {
    if (threadIdx.x < 128) s[threadIdx.x] = 0.f;
}

// ---------------- 1: global average pool  pooled[n*64+c] ----------------
__global__ __launch_bounds__(256) void pool_k(const float* __restrict__ x, float* __restrict__ pooled) {
    __shared__ float red[256];
    int b = blockIdx.x;                       // n*64 + c
    const float4* p = (const float4*)(x + (size_t)b * PLANE_IN);
    float s = 0.f;
    for (int i = threadIdx.x; i < PLANE_IN / 4; i += 256) {
        float4 v = p[i];
        s += (v.x + v.y) + (v.z + v.w);
    }
    red[threadIdx.x] = s;
    __syncthreads();
    for (int off = 128; off > 0; off >>= 1) {
        if (threadIdx.x < off) red[threadIdx.x] += red[threadIdx.x + off];
        __syncthreads();
    }
    if (threadIdx.x == 0) pooled[b] = red[0] * (1.0f / (float)PLANE_IN);
}

// ---------------- 2: attention (sigmoid of 1x1 conv on pooled) ----------------
// att[0..31] = v (n*4+k), att[32..63] = h
__global__ void att_k(const float* __restrict__ pooled,
                      const float* __restrict__ awv, const float* __restrict__ abv,
                      const float* __restrict__ awh, const float* __restrict__ abh,
                      float* __restrict__ att) {
    int t = threadIdx.x;
    if (t >= 64) return;
    int isH = t >> 5, r = t & 31, n = r >> 2, k = r & 3;
    const float* aw = isH ? awh : awv;
    const float* ab = isH ? abh : abv;
    const float* pc = pooled + n * 64 + (isH ? 32 : 0);
    float z = ab[k];
    for (int c = 0; c < 32; ++c) z += pc[c] * aw[k * 32 + c];
    att[t] = 1.0f / (1.0f + __expf(-z));
}

// ---------------- 3: weight prep into WMMA bf16 A-operand layout ----------------
// A 16x32 bf16 k order used here (B uses the same): lane<16 -> K {0..15}; lane>=16 -> K {16..31}
__device__ __forceinline__ int klocal0(int vv, int half) {
    return 16 * half + 2 * vv;
}

// wsq: [ct(4)][j(18)][lane(32)][vv(8)] dwords, k = tap*64+ci, tap = kh*3+kw
// wv/wh: [n(8)][ct(2)][j(3)][lane(32)][vv(8)] dwords, k = tap*32+ci
__global__ __launch_bounds__(256) void wprep_k(const float* __restrict__ sq_w,
                                               const float* __restrict__ w_v,
                                               const float* __restrict__ w_h,
                                               const float* __restrict__ att,
                                               unsigned int* __restrict__ wsq,
                                               unsigned int* __restrict__ wv,
                                               unsigned int* __restrict__ wh) {
    int idx = blockIdx.x * 256 + threadIdx.x;
    if (idx < 18432) {                                   // square conv weights
        int vv = idx & 7, t2 = idx >> 3, l = t2 & 31, t3 = t2 >> 5;
        int j = t3 % 18, ct = t3 / 18;
        int m = ct * 16 + (l & 15), half = l >> 4;
        int k0 = j * 32 + klocal0(vv, half);
        int tap = k0 >> 6, ci = k0 & 63;                 // tap constant within a 32-chunk
        float f0 = sq_w[(m * 64 + ci) * 9 + tap];
        float f1 = sq_w[(m * 64 + ci + 1) * 9 + tap];
        wsq[idx] = (unsigned int)f2bf(f0) | ((unsigned int)f2bf(f1) << 16);
    } else if (idx < 18432 + 12288) {                    // vertical CondConv aggregated
        int id = idx - 18432;
        int vv = id & 7, t2 = id >> 3, l = t2 & 31, t3 = t2 >> 5;
        int j = t3 % 3, t4 = t3 / 3, ct = t4 & 1, n = t4 >> 1;
        int m = ct * 16 + (l & 15), half = l >> 4;
        int ci = klocal0(vv, half);                      // kh == j
        float f0 = 0.f, f1 = 0.f;
        for (int k = 0; k < 4; ++k) {
            float a = att[n * 4 + k];
            f0 += a * w_v[((k * 32 + m) * 32 + ci) * 3 + j];
            f1 += a * w_v[((k * 32 + m) * 32 + ci + 1) * 3 + j];
        }
        wv[id] = (unsigned int)f2bf(f0) | ((unsigned int)f2bf(f1) << 16);
    } else if (idx < 18432 + 2 * 12288) {                // horizontal CondConv aggregated
        int id = idx - 18432 - 12288;
        int vv = id & 7, t2 = id >> 3, l = t2 & 31, t3 = t2 >> 5;
        int j = t3 % 3, t4 = t3 / 3, ct = t4 & 1, n = t4 >> 1;
        int m = ct * 16 + (l & 15), half = l >> 4;
        int ci = klocal0(vv, half);                      // kw == j
        float f0 = 0.f, f1 = 0.f;
        for (int k = 0; k < 4; ++k) {
            float a = att[32 + n * 4 + k];
            f0 += a * w_h[((k * 32 + m) * 32 + ci) * 3 + j];
            f1 += a * w_h[((k * 32 + m) * 32 + ci + 1) * 3 + j];
        }
        wh[id] = (unsigned int)f2bf(f0) | ((unsigned int)f2bf(f1) << 16);
    }
}

// ---------------- 4: 3x3/s2 conv 64->64 via WMMA, writes shuffled + bias ----------------
// LDS layout [row(3)][col(66)][ci padded to 72] -> a lane's 16-row B half is 16 contiguous
// bf16 = two aligned ds_load_b128. 28.5KB LDS.
#define SQ_CIP 72
__global__ __launch_bounds__(256) void sq_conv_k(const float* __restrict__ x,
                                                 const unsigned int* __restrict__ wA,
                                                 const float* __restrict__ bias,
                                                 float* __restrict__ out) {
    __shared__ unsigned short lx[3 * 66 * SQ_CIP];
    int bid = blockIdx.x;
    int owt = bid % 6; int t = bid / 6;
    int oh = t % OH;  int n = t / OH;
    int ow0 = owt * 32;
    int ih0 = 2 * oh - 1, iw0 = 2 * ow0 - 1;
    const float* xn = x + (size_t)n * CIN * PLANE_IN;

    for (int e = threadIdx.x; e < 64 * 3 * 66; e += 256) {
        int ci = e / 198; int rem = e - ci * 198;        // e = ci*198 + r*66 + c (coalesced in c)
        int r = rem / 66; int c = rem - r * 66;
        int ih = ih0 + r, iw = iw0 + c;
        float v = 0.f;
        if (ih >= 0 && ih < IHH && iw >= 0 && iw < IWW)
            v = xn[(size_t)ci * PLANE_IN + ih * IWW + iw];
        lx[(r * 66 + c) * SQ_CIP + ci] = f2bf(v);
    }
    __syncthreads();

    int lane = threadIdx.x & 31, wv = threadIdx.x >> 5;
    int ct = wv >> 1;                     // co tile 0..3
    int pxb = (wv & 1) << 4;              // pixel base 0/16
    int nn = lane & 15, half = lane >> 4;
    int px = pxb + nn;

    v8f acc = {};
    for (int j = 0; j < 18; ++j) {        // K = 9 taps * 64 ci = 576 = 18 * 32
        BfVec a;
        const uint4* ap = (const uint4*)(wA + ((((ct * 18 + j) << 5) + lane) << 3));
        a.u4[0] = ap[0];
        a.u4[1] = ap[1];

        int tap = j >> 1, cibase = (j & 1) << 5;
        int kh = tap / 3, kw = tap - 3 * kh;
        int col = 2 * px + kw;
        BfVec b;
        const uint4* bp = (const uint4*)(&lx[(kh * 66 + col) * SQ_CIP + cibase + (half << 4)]);
        b.u4[0] = bp[0];
        b.u4[1] = bp[1];

        acc = __builtin_amdgcn_wmma_f32_16x16x32_bf16(false, a.v, false, b.v,
                                                      (short)0, acc, false, false);
    }

    int ow = ow0 + px;
    #pragma unroll
    for (int r = 0; r < 8; ++r) {
        int co = ct * 16 + (half ? (8 + r) : r);
        int oc = shuffled(co);            // sq occupies concat channels 0..63
        out[((size_t)(n * 128 + oc) * OH + oh) * OW + ow] = acc[r] + bias[co];
    }
}

// ---------------- wave-level channel stat reduce + LDS/global atomics ----------------
__device__ __forceinline__ void stat_accum(const v8f& acc, int ct, int half, int nn,
                                           float* s_sum, float* s_ssq) {
    #pragma unroll
    for (int r = 0; r < 8; ++r) {
        int co = ct * 16 + (half ? (8 + r) : r);
        float s = acc[r], q = acc[r] * acc[r];
        #pragma unroll
        for (int m = 1; m <= 8; m <<= 1) {               // reduce the 16-lane half (one channel)
            s += __shfl_xor(s, m, 32);
            q += __shfl_xor(q, m, 32);
        }
        if (nn == 0) {
            atomicAdd(&s_sum[co], s);
            atomicAdd(&s_ssq[co], q);
        }
    }
}

// ---------------- 5: 3x1/s2 CondConv branch (channels 0..31) ----------------
#define V_CIP 40
__global__ __launch_bounds__(256) void v_conv_k(const float* __restrict__ x,
                                                const unsigned int* __restrict__ wA,
                                                float* __restrict__ out,
                                                float* __restrict__ gsum,
                                                float* __restrict__ gssq) {
    __shared__ unsigned short lx[3 * 64 * V_CIP];        // [row][px][ci pad 40], 15.4KB
    __shared__ float s_sum[32], s_ssq[32];
    int bid = blockIdx.x;
    int owt = bid % 3; int t = bid / 3;
    int oh = t % OH;  int n = t / OH;
    int ow0 = owt * 64;
    int ih0 = 2 * oh - 1;
    if (threadIdx.x < 32) { s_sum[threadIdx.x] = 0.f; s_ssq[threadIdx.x] = 0.f; }
    const float* xn = x + (size_t)n * CIN * PLANE_IN;    // x1: channels 0..31

    for (int e = threadIdx.x; e < 32 * 3 * 64; e += 256) {
        int px = e & 63; int t2 = e >> 6; int r = t2 % 3; int ci = t2 / 3;
        int ih = ih0 + r, iw = 2 * (ow0 + px);
        float v = (ih >= 0 && ih < IHH) ? xn[(size_t)ci * PLANE_IN + ih * IWW + iw] : 0.f;
        lx[(r * 64 + px) * V_CIP + ci] = f2bf(v);
    }
    __syncthreads();

    int lane = threadIdx.x & 31, wv = threadIdx.x >> 5;
    int ct = wv & 1;                      // co tile 0..1
    int pxb = (wv >> 1) << 4;             // pixel base 0..48
    int nn = lane & 15, half = lane >> 4;
    int px = pxb + nn;

    v8f acc = {};
    #pragma unroll
    for (int j = 0; j < 3; ++j) {         // K = 3 taps(kh) * 32 ci = 96
        BfVec a;
        const uint4* ap = (const uint4*)(wA + (((((n * 2 + ct) * 3 + j) << 5) + lane) << 3));
        a.u4[0] = ap[0];
        a.u4[1] = ap[1];
        BfVec b;
        const uint4* bp = (const uint4*)(&lx[(j * 64 + px) * V_CIP + (half << 4)]);
        b.u4[0] = bp[0];
        b.u4[1] = bp[1];
        acc = __builtin_amdgcn_wmma_f32_16x16x32_bf16(false, a.v, false, b.v,
                                                      (short)0, acc, false, false);
    }

    int ow = ow0 + px;
    #pragma unroll
    for (int r = 0; r < 8; ++r) {
        int co = ct * 16 + (half ? (8 + r) : r);
        int oc = shuffled(64 + co);       // v occupies concat channels 64..95
        out[((size_t)(n * 128 + oc) * OH + oh) * OW + ow] = acc[r];
    }
    stat_accum(acc, ct, half, nn, s_sum, s_ssq);
    __syncthreads();
    if (threadIdx.x < 32) {
        atomicAdd(&gsum[threadIdx.x], s_sum[threadIdx.x]);
        atomicAdd(&gssq[threadIdx.x], s_ssq[threadIdx.x]);
    }
}

// ---------------- 6: 1x3/s2 CondConv branch (channels 32..63) ----------------
__global__ __launch_bounds__(256) void h_conv_k(const float* __restrict__ x,
                                                const unsigned int* __restrict__ wA,
                                                float* __restrict__ out,
                                                float* __restrict__ gsum,
                                                float* __restrict__ gssq) {
    __shared__ unsigned short lx[130 * V_CIP];           // [col(129 used)][ci pad 40], 10.4KB
    __shared__ float s_sum[32], s_ssq[32];
    int bid = blockIdx.x;
    int owt = bid % 3; int t = bid / 3;
    int oh = t % OH;  int n = t / OH;
    int ow0 = owt * 64;
    int ih = 2 * oh;                      // kernel 1x3, pad(0,1): row always in range
    if (threadIdx.x < 32) { s_sum[threadIdx.x] = 0.f; s_ssq[threadIdx.x] = 0.f; }
    const float* xn = x + (size_t)n * CIN * PLANE_IN + (size_t)32 * PLANE_IN;  // x2

    for (int e = threadIdx.x; e < 32 * 129; e += 256) {
        int c = e % 129; int ci = e / 129;
        int iw = 2 * ow0 - 1 + c;
        float v = (iw >= 0 && iw < IWW) ? xn[(size_t)ci * PLANE_IN + ih * IWW + iw] : 0.f;
        lx[c * V_CIP + ci] = f2bf(v);
    }
    __syncthreads();

    int lane = threadIdx.x & 31, wv = threadIdx.x >> 5;
    int ct = wv & 1;
    int pxb = (wv >> 1) << 4;
    int nn = lane & 15, half = lane >> 4;
    int px = pxb + nn;

    v8f acc = {};
    #pragma unroll
    for (int j = 0; j < 3; ++j) {         // K = 3 taps(kw) * 32 ci = 96
        BfVec a;
        const uint4* ap = (const uint4*)(wA + (((((n * 2 + ct) * 3 + j) << 5) + lane) << 3));
        a.u4[0] = ap[0];
        a.u4[1] = ap[1];
        BfVec b;
        const uint4* bp = (const uint4*)(&lx[(2 * px + j) * V_CIP + (half << 4)]);
        b.u4[0] = bp[0];
        b.u4[1] = bp[1];
        acc = __builtin_amdgcn_wmma_f32_16x16x32_bf16(false, a.v, false, b.v,
                                                      (short)0, acc, false, false);
    }

    int ow = ow0 + px;
    #pragma unroll
    for (int r = 0; r < 8; ++r) {
        int co = ct * 16 + (half ? (8 + r) : r);
        int oc = shuffled(96 + co);       // h occupies concat channels 96..127
        out[((size_t)(n * 128 + oc) * OH + oh) * OW + ow] = acc[r];
    }
    stat_accum(acc, ct, half, nn, s_sum, s_ssq);
    __syncthreads();
    if (threadIdx.x < 32) {
        atomicAdd(&gsum[threadIdx.x], s_sum[threadIdx.x]);
        atomicAdd(&gssq[threadIdx.x], s_ssq[threadIdx.x]);
    }
}

// ---------------- 7: BN finalize: per-channel scale/shift from batch stats ----------------
__global__ void bn_finalize_k(const float* __restrict__ stats,
                              const float* __restrict__ gv, const float* __restrict__ bv,
                              const float* __restrict__ gh, const float* __restrict__ bh,
                              float* __restrict__ sc) {
    int t = threadIdx.x;
    if (t >= 64) return;
    int isH = t >> 5, c = t & 31;
    float s = stats[isH * 64 + c];
    float q = stats[isH * 64 + 32 + c];
    const float cnt = (float)NB * (float)PLANE_OUT;      // biased variance over (N,H,W)
    float m   = s / cnt;
    float var = q / cnt - m * m;
    float g = isH ? gh[c] : gv[c];
    float b = isH ? bh[c] : bv[c];
    float scale = g * rsqrtf(var + 1e-5f);
    sc[isH * 64 + c]      = scale;
    sc[isH * 64 + 32 + c] = b - m * scale;
}

// ---------------- 8: in-place BN affine on v/h planes of the shuffled output ----------------
__global__ __launch_bounds__(256) void bn_apply_k(float* __restrict__ out,
                                                  const float* __restrict__ sc) {
    int b = blockIdx.x;                   // n*64 + ch ; ch<32 -> v, else h
    int n = b >> 6, ch = b & 63;
    int isH = ch >> 5, co = ch & 31;
    float scale = sc[isH * 64 + co];
    float shift = sc[isH * 64 + 32 + co];
    int oc = shuffled((isH ? 96 : 64) + co);
    float4* p = (float4*)(out + (size_t)(n * 128 + oc) * PLANE_OUT);
    for (int i = threadIdx.x; i < PLANE_OUT / 4; i += 256) {
        float4 v = p[i];
        v.x = v.x * scale + shift;
        v.y = v.y * scale + shift;
        v.z = v.z * scale + shift;
        v.w = v.w * scale + shift;
        p[i] = v;
    }
}

// ---------------- launch ----------------
extern "C" void kernel_launch(void* const* d_in, const int* in_sizes, int n_in,
                              void* d_out, int out_size, void* d_ws, size_t ws_size,
                              hipStream_t stream) {
    (void)in_sizes; (void)n_in; (void)out_size; (void)ws_size;
    const float* x       = (const float*)d_in[0];
    const float* sq_w    = (const float*)d_in[1];
    const float* sq_b    = (const float*)d_in[2];
    const float* att_w_v = (const float*)d_in[3];
    const float* att_b_v = (const float*)d_in[4];
    const float* w_v     = (const float*)d_in[5];
    const float* bn_v_g  = (const float*)d_in[6];
    const float* bn_v_b  = (const float*)d_in[7];
    const float* att_w_h = (const float*)d_in[8];
    const float* att_b_h = (const float*)d_in[9];
    const float* w_h     = (const float*)d_in[10];
    const float* bn_h_g  = (const float*)d_in[11];
    const float* bn_h_b  = (const float*)d_in[12];
    float* out = (float*)d_out;

    // ws map (dwords): pooled[512] | att[64] | wA_sq[18432] | wA_v[12288] | wA_h[12288]
    //                  | stats[128] | scale_shift[128]  => ~176KB
    float*        ws     = (float*)d_ws;
    float*        pooled = ws;
    float*        att    = ws + 512;
    unsigned int* wA_sq  = (unsigned int*)(ws + 576);
    unsigned int* wA_v   = (unsigned int*)(ws + 19008);
    unsigned int* wA_h   = (unsigned int*)(ws + 31296);
    float*        stats  = ws + 43584;    // sum_v,ssq_v,sum_h,ssq_h (32 each)
    float*        scsh   = ws + 43712;    // scale_v,shift_v,scale_h,shift_h

    zero_stats_k<<<1, 128, 0, stream>>>(stats);
    pool_k<<<NB * CIN, 256, 0, stream>>>(x, pooled);
    att_k<<<1, 64, 0, stream>>>(pooled, att_w_v, att_b_v, att_w_h, att_b_h, att);
    wprep_k<<<168, 256, 0, stream>>>(sq_w, w_v, w_h, att, wA_sq, wA_v, wA_h);
    sq_conv_k<<<NB * OH * 6, 256, 0, stream>>>(x, wA_sq, sq_b, out);
    v_conv_k<<<NB * OH * 3, 256, 0, stream>>>(x, wA_v, out, stats, stats + 32);
    h_conv_k<<<NB * OH * 3, 256, 0, stream>>>(x, wA_h, out, stats + 64, stats + 96);
    bn_finalize_k<<<1, 64, 0, stream>>>(stats, bn_v_g, bn_v_b, bn_h_g, bn_h_b, scsh);
    bn_apply_k<<<NB * 64, 256, 0, stream>>>(out, scsh);
}
